// SmyrfAttention_8976481649325
// MI455X (gfx1250) — matrix operand order repaired and checked by reading, compiled, not run on hardware
//
#include <hip/hip_runtime.h>
#include <hip/hip_bf16.h>
#include <math.h>

// ---------------------------------------------------------------------------
// SMYRF attention for MI455X (gfx1250).
//   B=4, N=4096, H=8, D=64, BUCKET=128, N_HASHES=4
// Phases: hash -> bitonic argsort -> bucketed WMMA attention -> combine+proj.
// ---------------------------------------------------------------------------

#define BB 4
#define NN 4096
#define HH 8
#define DD 64
#define HD 512           // H*D
#define BHN 32           // B*H
#define NHASH 4
#define BUCKET 128
#define NB 32            // N / BUCKET
#define SLOTS 128        // NHASH * BHN

typedef __attribute__((ext_vector_type(16))) _Float16 v16h;
typedef __attribute__((ext_vector_type(8)))  float    v8f;
typedef __attribute__((ext_vector_type(4)))  int      v4i;

#if defined(__has_builtin)
#if __has_builtin(__builtin_amdgcn_global_load_async_to_lds_b128) && \
    __has_builtin(__builtin_amdgcn_s_wait_asynccnt)
#define SMYRF_ASYNC 1
typedef __attribute__((address_space(1))) v4i* gptr4;
typedef __attribute__((address_space(3))) v4i* lptr4;
#endif
#endif

// ---------------- WMMA fragment loaders (wave32; A and B use contiguous k = 16*(lane/16) + s) -----

// A fragment: 16x32 f16, source stored row-major [m][k] with given stride.
__device__ static inline v16h load_a_frag_h(const _Float16* base, int stride,
                                            int row0, int k0, int lane) {
    int m = row0 + (lane & 15);
    int khalf = (lane >> 4) * 16;
    const _Float16* p = base + m * stride + k0;
    v16h a;
#pragma unroll
    for (int i = 0; i < 4; ++i) {           // VGPR 0..3 : K = khalf + 2i, +1
        int kk = khalf + 2 * i;
        a[2 * i]     = p[kk];
        a[2 * i + 1] = p[kk + 1];
    }
#pragma unroll
    for (int i = 4; i < 8; ++i) {           // VGPR 4..7 : K = khalf + 8 + ...
        int kk = khalf + 8 + 2 * (i - 4);
        a[2 * i]     = p[kk];
        a[2 * i + 1] = p[kk + 1];
    }
    return a;
}

// B fragment 32x16 where the source matrix is stored [n][k] (i.e. B = X^T),
// so per lane (n fixed) the K run is contiguous -> 2x ds_load_b128.
__device__ static inline v16h load_b_frag_nk(const _Float16* base, int stride,
                                             int n0, int k0, int lane) {
    int n = n0 + (lane & 15);
    int kb = (lane >> 4) * 16;
    const _Float16* p = base + n * stride + k0 + kb;
    v16h b;
#pragma unroll
    for (int i = 0; i < 16; ++i) b[i] = p[i];
    return b;
}

__device__ static inline v8f zero8() {
    v8f z = {0.f, 0.f, 0.f, 0.f, 0.f, 0.f, 0.f, 0.f};
    return z;
}

// ---------------- Kernel 1: norms + E2LSH hashes ---------------------------
// grid = BH (32), block = 256
__global__ __launch_bounds__(256) void smyrf_hash_kernel(
    const float* __restrict__ q, const float* __restrict__ k,
    const float* __restrict__ alpha, const float* __restrict__ beta,
    float* __restrict__ qh, float* __restrict__ kh) {
    int bh = blockIdx.x;
    int b = bh >> 3, h = bh & 7;
    int t = threadIdx.x;

    __shared__ float qn2[NN];
    __shared__ float kn2[NN];
    __shared__ float redq[256], redk[256];
    __shared__ float M2s;

    float mq = 0.f, mk = 0.f;
#pragma unroll 1
    for (int i = 0; i < NN / 256; ++i) {
        int n = t + i * 256;
        const float4* qr = (const float4*)(q + ((size_t)(b * NN + n)) * HD + h * DD);
        const float4* kr = (const float4*)(k + ((size_t)(b * NN + n)) * HD + h * DD);
        float sq = 0.f, sk = 0.f;
#pragma unroll
        for (int d = 0; d < DD / 4; ++d) {
            float4 a = qr[d];
            sq += a.x * a.x + a.y * a.y + a.z * a.z + a.w * a.w;
            float4 c = kr[d];
            sk += c.x * c.x + c.y * c.y + c.z * c.z + c.w * c.w;
        }
        qn2[n] = sq; kn2[n] = sk;
        mq = fmaxf(mq, sq); mk = fmaxf(mk, sk);
    }
    redq[t] = mq; redk[t] = mk;
    __syncthreads();
    for (int s = 128; s > 0; s >>= 1) {
        if (t < s) {
            redq[t] = fmaxf(redq[t], redq[t + s]);
            redk[t] = fmaxf(redk[t], redk[t + s]);
        }
        __syncthreads();
    }
    if (t == 0) M2s = redq[0] + redk[0];   // MQ^2 + MK^2
    __syncthreads();
    float M2 = M2s;

#pragma unroll 1
    for (int i = 0; i < NN / 256; ++i) {
        int n = t + i * 256;
        const float* qr = q + ((size_t)(b * NN + n)) * HD + h * DD;
        const float* kr = k + ((size_t)(b * NN + n)) * HD + h * DD;
        float extq = sqrtf(fmaxf(M2 - qn2[n], 0.f));
        float extk = sqrtf(fmaxf(M2 - kn2[n], 0.f));
        float aq[NHASH] = {0.f, 0.f, 0.f, 0.f};
        float ak[NHASH] = {0.f, 0.f, 0.f, 0.f};
        for (int d = 0; d < DD; ++d) {
            float qv = qr[d], kv = kr[d];
#pragma unroll
            for (int j = 0; j < NHASH; ++j) {
                float al = alpha[d * NHASH + j];
                aq[j] += qv * al;
                ak[j] += kv * al;
            }
        }
#pragma unroll
        for (int j = 0; j < NHASH; ++j) {
            // q_t = [q, ext_q, 0] -> alpha row D;  k_t = [k, 0, ext_k] -> row D+1
            qh[((size_t)(j * BHN + bh)) * NN + n] =
                aq[j] + extq * alpha[DD * NHASH + j] + beta[j];
            kh[((size_t)(j * BHN + bh)) * NN + n] =
                ak[j] + extk * alpha[(DD + 1) * NHASH + j] + beta[j];
        }
    }
}

// ---------------- Kernel 2: bitonic argsort of 4096 keys -------------------
// grid = SLOTS (128), block = 1024
__global__ __launch_bounds__(1024) void smyrf_sort_kernel(
    const float* __restrict__ hashes, int* __restrict__ pos) {
    int slot = blockIdx.x;
    const float* src = hashes + (size_t)slot * NN;
    __shared__ float key[NN];
    __shared__ int   idx[NN];
    int t = threadIdx.x;

#ifdef SMYRF_ASYNC
    // contiguous global -> LDS copy, no conversion: use the async-to-LDS path
    // (ASYNCcnt-tracked, bypasses VGPRs). One b128 per thread covers 4096 f32.
    __builtin_amdgcn_global_load_async_to_lds_b128(
        (gptr4)(src + 4 * t), (lptr4)(key + 4 * t), 0, 0);
    for (int i = t; i < NN; i += 1024) idx[i] = i;
    __builtin_amdgcn_s_wait_asynccnt(0);
#else
    for (int i = t; i < NN; i += 1024) { key[i] = src[i]; idx[i] = i; }
#endif
    __syncthreads();

    for (int ksz = 2; ksz <= NN; ksz <<= 1) {
        for (int j = ksz >> 1; j > 0; j >>= 1) {
            for (int i = t; i < NN; i += 1024) {
                int ixj = i ^ j;
                if (ixj > i) {
                    bool asc = ((i & ksz) == 0);
                    float a = key[i], c = key[ixj];
                    bool sw = asc ? (a > c) : (a < c);
                    if (sw) {
                        key[i] = c; key[ixj] = a;
                        int ti = idx[i]; idx[i] = idx[ixj]; idx[ixj] = ti;
                    }
                }
            }
            __syncthreads();
        }
    }
    for (int i = t; i < NN; i += 1024) pos[(size_t)slot * NN + i] = idx[i];
}

// ---------------- Kernel 3: bucketed attention (WMMA) ----------------------
// grid = SLOTS * NB (4096), block = 256 (8 wave32)
__global__ __launch_bounds__(256) void smyrf_attn_kernel(
    const float* __restrict__ q, const float* __restrict__ k,
    const float* __restrict__ v, const int* __restrict__ q_pos,
    const int* __restrict__ k_pos, _Float16* __restrict__ o,
    float* __restrict__ lse_out) {
    int bucket = blockIdx.x & (NB - 1);
    int hb = blockIdx.x >> 5;                // j * 32 + bh
    int bh = hb & (BHN - 1);
    int b = bh >> 3, h = bh & 7;

    __shared__ _Float16 Qh[BUCKET][DD + 8];      // [m][k]
    __shared__ _Float16 Kh[BUCKET][DD + 8];      // [n][k] for GEMM1 B
    __shared__ _Float16 Vt[DD][BUCKET + 8];      // [n][k] for GEMM2 B (V^T)
    __shared__ float    S[BUCKET][BUCKET + 2];   // raw scores
    __shared__ _Float16 P[BUCKET][BUCKET + 8];   // normalized probs (f16)
    __shared__ int      qidx[BUCKET], kidx[BUCKET];

    int t = threadIdx.x;
    int lane = t & 31;
    int w = t >> 5;

    if (t < BUCKET) {
        qidx[t] = q_pos[(size_t)hb * NN + bucket * BUCKET + t];
        kidx[t] = k_pos[(size_t)hb * NN + bucket * BUCKET + t];
    }
    __syncthreads();

    // gather sorted q/k/v rows into LDS as f16: 2 threads per row, float4 loads
    {
        int r = t >> 1;
        int c0 = (t & 1) * 32;
        const float4* qr = (const float4*)(q + ((size_t)(b * NN + qidx[r])) * HD + h * DD + c0);
        const float4* kr = (const float4*)(k + ((size_t)(b * NN + kidx[r])) * HD + h * DD + c0);
        const float4* vr = (const float4*)(v + ((size_t)(b * NN + kidx[r])) * HD + h * DD + c0);
#pragma unroll
        for (int i = 0; i < 8; ++i) {
            float4 qv = qr[i], kv = kr[i], vv = vr[i];
            int c = c0 + 4 * i;
            Qh[r][c]     = (_Float16)qv.x;
            Qh[r][c + 1] = (_Float16)qv.y;
            Qh[r][c + 2] = (_Float16)qv.z;
            Qh[r][c + 3] = (_Float16)qv.w;
            Kh[r][c]     = (_Float16)kv.x;
            Kh[r][c + 1] = (_Float16)kv.y;
            Kh[r][c + 2] = (_Float16)kv.z;
            Kh[r][c + 3] = (_Float16)kv.w;
            Vt[c][r]     = (_Float16)vv.x;      // transposed store
            Vt[c + 1][r] = (_Float16)vv.y;
            Vt[c + 2][r] = (_Float16)vv.z;
            Vt[c + 3][r] = (_Float16)vv.w;
        }
    }
    __syncthreads();

    // ---- GEMM1: S = (Q K^T) * temp ; each wave owns 16 rows x 128 cols ----
    const float temp = 0.125f;   // 1/sqrt(D)
    int wrow = w * 16;
    v8f acc[8];
#pragma unroll
    for (int nt = 0; nt < 8; ++nt) acc[nt] = zero8();
#pragma unroll
    for (int kt = 0; kt < 2; ++kt) {
        v16h a = load_a_frag_h(&Qh[0][0], DD + 8, wrow, kt * 32, lane);
#pragma unroll
        for (int nt = 0; nt < 8; ++nt) {
            v16h bm = load_b_frag_nk(&Kh[0][0], DD + 8, nt * 16, kt * 32, lane);
            acc[nt] = __builtin_amdgcn_wmma_f32_16x16x32_f16(
                false, a, false, bm, (short)0, acc[nt], false, false);
        }
    }
    int mb = (lane >> 4) * 8;
    int nn = lane & 15;
#pragma unroll
    for (int nt = 0; nt < 8; ++nt)
#pragma unroll
        for (int vv = 0; vv < 8; ++vv)
            S[wrow + mb + vv][nt * 16 + nn] = acc[nt][vv] * temp;
    __syncthreads();

    // ---- row softmax: write normalized probs as f16 into P, keep lse ----
    if (t < BUCKET) {
        float m = -1e30f;
        for (int j2 = 0; j2 < BUCKET; ++j2) m = fmaxf(m, S[t][j2]);
        float sum = 0.f;
        float e[BUCKET];
#pragma unroll 1
        for (int j2 = 0; j2 < BUCKET; ++j2) {
            float ev = __expf(S[t][j2] - m);
            e[j2] = ev;
            sum += ev;
        }
        float inv = 1.f / sum;
#pragma unroll 1
        for (int j2 = 0; j2 < BUCKET; ++j2) P[t][j2] = (_Float16)(e[j2] * inv);
        lse_out[(size_t)hb * NN + qidx[t]] = m + __logf(sum);
    }
    __syncthreads();

    // ---- GEMM2: O = P @ V ; 16 rows x 64 cols per wave, K = 128 ----
    v8f oacc[4];
#pragma unroll
    for (int nt = 0; nt < 4; ++nt) oacc[nt] = zero8();
#pragma unroll
    for (int kt = 0; kt < 4; ++kt) {
        v16h a = load_a_frag_h(&P[0][0], BUCKET + 8, wrow, kt * 32, lane);
#pragma unroll
        for (int nt = 0; nt < 4; ++nt) {
            v16h bm = load_b_frag_nk(&Vt[0][0], BUCKET + 8, nt * 16, kt * 32, lane);
            oacc[nt] = __builtin_amdgcn_wmma_f32_16x16x32_f16(
                false, a, false, bm, (short)0, oacc[nt], false, false);
        }
    }
    // scatter through q permutation (unsort fused here)
#pragma unroll
    for (int nt = 0; nt < 4; ++nt)
#pragma unroll
        for (int vv = 0; vv < 8; ++vv) {
            int r = wrow + mb + vv;
            int col = nt * 16 + nn;
            o[((size_t)hb * NN + qidx[r]) * DD + col] = (_Float16)oacc[nt][vv];
        }
}

// ---------------- Kernel 4: hash-combine + output projection ---------------
// grid = B * (N/64) = 256, block = 256 (8 wave32); 64 tokens per block
__global__ __launch_bounds__(256) void smyrf_combine_proj_kernel(
    const _Float16* __restrict__ o, const float* __restrict__ lse,
    const float* __restrict__ out_w, const float* __restrict__ out_b,
    float* __restrict__ out) {
    int blk = blockIdx.x;
    int b = blk >> 6;
    int n0 = (blk & 63) * 64;

    __shared__ _Float16 comb[64][HD + 8];   // [m][k]
    __shared__ _Float16 wl[DD][HD + 8];     // [n][k] : wl[dout][din] = W^T tile
    __shared__ float    probs[64][HH][NHASH];

    int t = threadIdx.x;
    int lane = t & 31;
    int w = t >> 5;

    // per-(token, head) softmax over the 4 hash-round lse values
    for (int p = t; p < 64 * HH; p += 256) {
        int tk = p >> 3, h = p & 7;
        int bh = b * HH + h;
        int n = n0 + tk;
        float l0 = lse[((size_t)(0 * BHN + bh)) * NN + n];
        float l1 = lse[((size_t)(1 * BHN + bh)) * NN + n];
        float l2 = lse[((size_t)(2 * BHN + bh)) * NN + n];
        float l3 = lse[((size_t)(3 * BHN + bh)) * NN + n];
        float m = fmaxf(fmaxf(l0, l1), fmaxf(l2, l3));
        float e0 = __expf(l0 - m), e1 = __expf(l1 - m);
        float e2 = __expf(l2 - m), e3 = __expf(l3 - m);
        float inv = 1.f / (e0 + e1 + e2 + e3);
        probs[tk][h][0] = e0 * inv;
        probs[tk][h][1] = e1 * inv;
        probs[tk][h][2] = e2 * inv;
        probs[tk][h][3] = e3 * inv;
    }
    // stage out_w (row-major [dout][din]) directly: contiguous load + store
    for (int p = t; p < DD * HD; p += 256) {
        int dout = p >> 9, din = p & (HD - 1);
        wl[dout][din] = (_Float16)out_w[p];
    }
    __syncthreads();

    // combined hidden = sum_j probs * o_j, laid out (token, h*64+d)
    for (int p = t; p < 64 * HD; p += 256) {
        int tk = p >> 9, c = p & (HD - 1);
        int h = c >> 6, d = c & 63;
        int bh = b * HH + h;
        int n = n0 + tk;
        float a = 0.f;
#pragma unroll
        for (int j = 0; j < NHASH; ++j)
            a += probs[tk][h][j] *
                 (float)o[((size_t)(j * BHN + bh) * NN + n) * DD + d];
        comb[tk][c] = (_Float16)a;
    }
    __syncthreads();

    // ---- WMMA GEMM: (64 x 512) @ (512 x 64) ----
    int strip = w & 3;            // token strip (16 rows)
    int nbase = (w >> 2) * 32;    // output-col half
    v8f acc[2];
    acc[0] = zero8(); acc[1] = zero8();
#pragma unroll 1
    for (int kt = 0; kt < HD / 32; ++kt) {
        v16h a = load_a_frag_h(&comb[0][0], HD + 8, strip * 16, kt * 32, lane);
#pragma unroll
        for (int nt = 0; nt < 2; ++nt) {
            v16h bm = load_b_frag_nk(&wl[0][0], HD + 8, nbase + nt * 16, kt * 32, lane);
            acc[nt] = __builtin_amdgcn_wmma_f32_16x16x32_f16(
                false, a, false, bm, (short)0, acc[nt], false, false);
        }
    }
    int mb = (lane >> 4) * 8;
    int nn = lane & 15;
#pragma unroll
    for (int nt = 0; nt < 2; ++nt)
#pragma unroll
        for (int vv = 0; vv < 8; ++vv) {
            int tk = strip * 16 + mb + vv;
            int dout = nbase + nt * 16 + nn;
            out[((size_t)(b * NN + n0 + tk)) * DD + dout] =
                acc[nt][vv] + out_b[dout];
        }
}

// ---------------------------------------------------------------------------
extern "C" void kernel_launch(void* const* d_in, const int* in_sizes, int n_in,
                              void* d_out, int out_size, void* d_ws, size_t ws_size,
                              hipStream_t stream) {
    const float* query = (const float*)d_in[0];
    const float* key_i = (const float*)d_in[1];
    const float* value = (const float*)d_in[2];
    const float* alpha = (const float*)d_in[3];
    const float* beta  = (const float*)d_in[4];
    const float* out_w = (const float*)d_in[5];
    const float* out_b = (const float*)d_in[6];
    float* out = (float*)d_out;

    const size_t slotN = (size_t)SLOTS * NN;          // 524288
    float* qh    = (float*)d_ws;
    float* kh    = qh + slotN;
    int*   q_pos = (int*)(kh + slotN);
    int*   k_pos = q_pos + slotN;
    _Float16* o  = (_Float16*)(k_pos + slotN);        // SLOTS*N*D halves (64 MB)
    float* lse   = (float*)(o + slotN * DD);

    smyrf_hash_kernel<<<BHN, 256, 0, stream>>>(query, key_i, alpha, beta, qh, kh);
    smyrf_sort_kernel<<<SLOTS, 1024, 0, stream>>>(qh, q_pos);
    smyrf_sort_kernel<<<SLOTS, 1024, 0, stream>>>(kh, k_pos);
    smyrf_attn_kernel<<<SLOTS * NB, 256, 0, stream>>>(query, key_i, value,
                                                      q_pos, k_pos, o, lse);
    smyrf_combine_proj_kernel<<<BB * (NN / 64), 256, 0, stream>>>(o, lse, out_w,
                                                                  out_b, out);
}
